// CompositionalNetwork_73839077753097
// MI455X (gfx1250) — matrix operand order repaired and checked by reading, not compiled, not run on hardware
//
#include <hip/hip_runtime.h>

// ---------------------------------------------------------------------------
// CompositionalNetwork on MI455X (gfx1250, wave32, WMMA)
//
// Y_k[C,200] = gather(X)[C,220k] @ W_k^T + b_k   for k = 1..4, C = 40000
// Split-precision bf16 GEMM: v ~ hi + lo (both bf16), P = Ah*Bh + Ah*Bl + Al*Bh
// accumulated in f32 via v_wmma_f32_16x16x32_bf16.
// ---------------------------------------------------------------------------

typedef __attribute__((ext_vector_type(16))) __bf16 v16bf;
typedef __attribute__((ext_vector_type(8)))  __bf16 v8bf;
typedef __attribute__((ext_vector_type(8)))  float  v8f;
typedef __attribute__((ext_vector_type(4)))  float  v4f;

#define CDIM   200      // comp_emb_dim (output cols)
#define EDIM   220      // per-token embedding dim
#define NCHUNK 40000    // chunks per length class
#define NPAD   208      // CDIM padded to multiple of 16

// ---------------------------------------------------------------------------
// Prep: split W_k (f32 [200 x 220k]) into zero-padded hi/lo bf16 [208 x 224k]
// ---------------------------------------------------------------------------
__global__ void prep_w_kernel(const float* __restrict__ W,
                              __bf16* __restrict__ hi, __bf16* __restrict__ lo,
                              int ksrc, int kpad) {
    const int total = NPAD * kpad;
    for (int i = blockIdx.x * blockDim.x + threadIdx.x; i < total;
         i += gridDim.x * blockDim.x) {
        const int r = i / kpad;
        const int c = i - r * kpad;
        float v = 0.0f;
        if (r < CDIM && c < ksrc) v = W[(size_t)r * ksrc + c];
        const __bf16 h = (__bf16)v;
        hi[i] = h;
        lo[i] = (__bf16)(v - (float)h);
    }
}

// ---------------------------------------------------------------------------
// Main fused gather + split + GEMM + bias + scatter kernel (one per k)
// ---------------------------------------------------------------------------
template <int KL>
__global__ void comp_kernel(const int*   __restrict__ tok,
                            const int*   __restrict__ tag,
                            const int*   __restrict__ starts,
                            const int*   __restrict__ pos,
                            const float* __restrict__ wordT,
                            const float* __restrict__ tagT,
                            const __bf16* __restrict__ Whi,
                            const __bf16* __restrict__ Wlo,
                            const float* __restrict__ bias,
                            float* __restrict__ out) {
    constexpr int KPAD  = 224 * KL;                 // K padded to mult of 32
    constexpr int KSRC  = 220 * KL;                 // real K
    constexpr int MT    = (KL <= 2) ? 32 : 16;      // chunks per workgroup
    constexpr int TILES = (MT / 16) * 13;           // 16x16 output tiles per WG

    extern __shared__ __bf16 smem[];
    __bf16* __restrict__ Xhi = smem;                // [MT][KPAD]
    __bf16* __restrict__ Xlo = smem + MT * KPAD;    // [MT][KPAD]

    const int tid = threadIdx.x;
    const int c0  = blockIdx.x * MT;                // first chunk of this WG

    // ---- Phase 1: gather embeddings for MT chunks, split f32 -> hi/lo bf16 --
    for (int i = tid; i < MT * KPAD; i += blockDim.x) {
        const int r = i / KPAD;
        const int c = i - r * KPAD;
        float v = 0.0f;
        if (c < KSRC) {
            const int j = c / EDIM;                 // token within chunk (< KL)
            const int e = c - j * EDIM;             // feature within token
            const int t = starts[c0 + r] + j;       // global token index
            if (e < 200) v = wordT[(size_t)tok[t] * 200 + e];
            else         v = tagT[(size_t)tag[t] * 20 + (e - 200)];
        }
        const __bf16 h = (__bf16)v;
        Xhi[i] = h;
        Xlo[i] = (__bf16)(v - (float)h);
    }
    __syncthreads();

    // ---- Phase 2: WMMA GEMM --------------------------------------------------
    const int wave  = tid >> 5;
    const int lane  = tid & 31;
    const int lhalf = lane >> 4;                    // 0: lanes 0-15, 1: 16-31
    const int lrow  = lane & 15;

    for (int tile = wave; tile < TILES; tile += 8) {
        const int mt    = tile / 13;                // m-tile (chunk block)
        const int nt    = tile - mt * 13;           // n-tile (output cols)
        const int nbase = nt * 16;

        // A operand = W rows (16 output cols x KPAD).  ISA 16-bit A layout:
        // lanes 0-15: M=lane, K in [k0,k0+8) and [k0+16,k0+24);
        // lanes 16-31 shifted by 8 in K.
        const __bf16* __restrict__ wh = Whi + (size_t)(nbase + lrow) * KPAD;
        const __bf16* __restrict__ wl = Wlo + (size_t)(nbase + lrow) * KPAD;
        // B operand = X rows (KPAD x 16 chunks).  Lanes 0-15: K in [k0,k0+8) and
        // [k0+16,k0+24) of chunk=lane; lanes 16-31 shifted by 8 in K.
        const __bf16* __restrict__ xh = Xhi + (mt * 16 + lrow) * KPAD;
        const __bf16* __restrict__ xl = Xlo + (mt * 16 + lrow) * KPAD;

        v8f acc = {};
        for (int kk = 0; kk < KPAD; kk += 32) {
            const int aoff = kk + lhalf * 8;
            const v8bf ah0 = *(const v8bf*)(wh + aoff);
            const v8bf ah1 = *(const v8bf*)(wh + aoff + 16);
            const v8bf al0 = *(const v8bf*)(wl + aoff);
            const v8bf al1 = *(const v8bf*)(wl + aoff + 16);
            const v16bf Ah = __builtin_shufflevector(
                ah0, ah1, 0, 1, 2, 3, 4, 5, 6, 7, 8, 9, 10, 11, 12, 13, 14, 15);
            const v16bf Al = __builtin_shufflevector(
                al0, al1, 0, 1, 2, 3, 4, 5, 6, 7, 8, 9, 10, 11, 12, 13, 14, 15);

            const int boff = kk + lhalf * 8;
            const v8bf bh0 = *(const v8bf*)(xh + boff);
            const v8bf bh1 = *(const v8bf*)(xh + boff + 16);
            const v8bf bl0 = *(const v8bf*)(xl + boff);
            const v8bf bl1 = *(const v8bf*)(xl + boff + 16);
            const v16bf Bh = __builtin_shufflevector(
                bh0, bh1, 0, 1, 2, 3, 4, 5, 6, 7, 8, 9, 10, 11, 12, 13, 14, 15);
            const v16bf Bl = __builtin_shufflevector(
                bl0, bl1, 0, 1, 2, 3, 4, 5, 6, 7, 8, 9, 10, 11, 12, 13, 14, 15);

            // hi*hi + hi*lo + lo*hi  (f32 accumulate; lo*lo term negligible)
            acc = __builtin_amdgcn_wmma_f32_16x16x32_bf16(
                false, Ah, false, Bh, (short)0, acc, false, false);
            acc = __builtin_amdgcn_wmma_f32_16x16x32_bf16(
                false, Ah, false, Bl, (short)0, acc, false, false);
            acc = __builtin_amdgcn_wmma_f32_16x16x32_bf16(
                false, Al, false, Bh, (short)0, acc, false, false);
        }

        // ---- Epilogue: bias + scatter.  D layout: VGPR j, lanes 0-15 ->
        // (col nbase+j, chunk=lane); lanes 16-31 -> (col nbase+8+j, chunk=lane-16).
        const int col0 = nbase + lhalf * 8;
        if (col0 < CDIM) {                          // mask padded cols 200..207
            const int row = pos[c0 + mt * 16 + lrow];
            float* __restrict__ o = out + (size_t)row * CDIM + col0;
            v4f o0 = {acc[0] + bias[col0 + 0], acc[1] + bias[col0 + 1],
                      acc[2] + bias[col0 + 2], acc[3] + bias[col0 + 3]};
            v4f o1 = {acc[4] + bias[col0 + 4], acc[5] + bias[col0 + 5],
                      acc[6] + bias[col0 + 6], acc[7] + bias[col0 + 7]};
            *(v4f*)(o + 0) = o0;                    // global_store_b128
            *(v4f*)(o + 4) = o1;                    // global_store_b128
        }
    }
}

// ---------------------------------------------------------------------------
// Launch
// ---------------------------------------------------------------------------
extern "C" void kernel_launch(void* const* d_in, const int* in_sizes, int n_in,
                              void* d_out, int out_size, void* d_ws, size_t ws_size,
                              hipStream_t stream) {
    (void)in_sizes; (void)n_in; (void)out_size; (void)ws_size;

    const int*   tok   = (const int*)d_in[0];
    const int*   tag   = (const int*)d_in[1];
    const int*   starts[4] = {(const int*)d_in[2], (const int*)d_in[4],
                              (const int*)d_in[6], (const int*)d_in[8]};
    const int*   pos[4]    = {(const int*)d_in[3], (const int*)d_in[5],
                              (const int*)d_in[7], (const int*)d_in[9]};
    const float* wordT = (const float*)d_in[10];
    const float* tagT  = (const float*)d_in[11];
    const float* W[4]  = {(const float*)d_in[12], (const float*)d_in[14],
                          (const float*)d_in[16], (const float*)d_in[18]};
    const float* b[4]  = {(const float*)d_in[13], (const float*)d_in[15],
                          (const float*)d_in[17], (const float*)d_in[19]};
    float* out = (float*)d_out;

    // Workspace layout: [W1hi][W1lo][W2hi][W2lo][W3hi][W3lo][W4hi][W4lo]
    // region k: NPAD * 224k bf16 elements each for hi and lo  (total ~1.8 MB)
    __bf16* wsb = (__bf16*)d_ws;
    __bf16 *whi[4], *wlo[4];
    size_t off = 0;
    for (int k = 1; k <= 4; ++k) {
        const size_t sz = (size_t)NPAD * 224 * k;
        whi[k - 1] = wsb + off; off += sz;
        wlo[k - 1] = wsb + off; off += sz;
    }

    // Split weights into hi/lo bf16 (once per call; small).
    for (int k = 1; k <= 4; ++k) {
        const int total = NPAD * 224 * k;
        prep_w_kernel<<<(total + 255) / 256, 256, 0, stream>>>(
            W[k - 1], whi[k - 1], wlo[k - 1], 220 * k, 224 * k);
    }

    // Fused gather + GEMM kernels, one per chunk length.
    comp_kernel<1><<<NCHUNK / 32, 256, 32 * 224 * 1 * 4, stream>>>(
        tok, tag, starts[0], pos[0], wordT, tagT, whi[0], wlo[0], b[0], out);
    comp_kernel<2><<<NCHUNK / 32, 256, 32 * 224 * 2 * 4, stream>>>(
        tok, tag, starts[1], pos[1], wordT, tagT, whi[1], wlo[1], b[1], out);
    comp_kernel<3><<<NCHUNK / 16, 256, 16 * 224 * 3 * 4, stream>>>(
        tok, tag, starts[2], pos[2], wordT, tagT, whi[2], wlo[2], b[2], out);
    comp_kernel<4><<<NCHUNK / 16, 256, 16 * 224 * 4 * 4, stream>>>(
        tok, tag, starts[3], pos[3], wordT, tagT, whi[3], wlo[3], b[3], out);
}
